// GCNet_23038204576098
// MI455X (gfx1250) — compile-verified, matrix operand order checked by reading, not run on hardware
//
#include <hip/hip_runtime.h>
#include <hip/hip_bf16.h>
#include <stdint.h>

typedef __bf16 bf16x16 __attribute__((ext_vector_type(16)));
typedef float  f32x8   __attribute__((ext_vector_type(8)));

// K-offset of element j (0..15) of a 16-bit 16x32 A/B fragment for lane-half h,
// per CDNA5 ISA 7.12.2. Note j=0..7 are contiguous K = 8h+0..7, j=8..15 are K = 16+8h+0..7.
__device__ __forceinline__ int frag_k(int j, int half) {
    int p  = j >> 1;
    int kb = ((p & 3) << 1) + ((p >> 2) << 4);
    return kb + (half << 3) + (j & 1);
}

__device__ __forceinline__ __bf16 to_bf16(float f) { return (__bf16)f; }

__device__ __forceinline__ float4 relu4(float4 v) {
    v.x = fmaxf(v.x, 0.0f); v.y = fmaxf(v.y, 0.0f);
    v.z = fmaxf(v.z, 0.0f); v.w = fmaxf(v.w, 0.0f);
    return v;
}

// ---------------- degree / normalization ----------------
__global__ void gcn_fill_deg(float* __restrict__ deg, int n) {
    int i = blockIdx.x * blockDim.x + threadIdx.x;
    if (i < n) deg[i] = 1.0f;                  // self-loop weight
}

__global__ void gcn_accum_deg(float* __restrict__ deg,
                              const long long* __restrict__ dst,
                              const float* __restrict__ ew, int E) {
    int e = blockIdx.x * blockDim.x + threadIdx.x;
    if (e < E) atomicAdd(&deg[(int)dst[e]], ew[e]);
}

__global__ void gcn_make_dinv(float* __restrict__ deg, int n) {
    int i = blockIdx.x * blockDim.x + threadIdx.x;
    if (i < n) deg[i] = rsqrtf(deg[i]);        // deg >= 1 always
}

// ---------------- pack W (f32, KxN row-major) into bf16 WMMA B-fragments ----------------
// One wave per (kblock, tn). Fragment block = 32 lanes * 16 bf16 = 1024 B, lane-major,
// so the GEMM reads each lane's whole fragment as one contiguous 32 B load.
__global__ void gcn_pack_w(const float* __restrict__ W, __bf16* __restrict__ Wpk,
                           int K, int Nout)
{
    int wave = blockIdx.x * (blockDim.x >> 5) + (threadIdx.x >> 5);
    int tiles_n = Nout >> 4;
    int kblocks = K >> 5;
    if (wave >= tiles_n * kblocks) return;
    int kb = wave / tiles_n;                   // matches GEMM index (kb*TILES_N + tn)
    int lane = threadIdx.x & 31;
    int half = lane >> 4;
    int col  = ((wave % tiles_n) << 4) + (lane & 15);
    __bf16* out = Wpk + ((size_t)wave << 9) + (lane << 4);
#pragma unroll
    for (int j = 0; j < 16; ++j) {
        int k = (kb << 5) + frag_k(j, half);
        out[j] = to_bf16(W[(size_t)k * Nout + col]);
    }
}

// ---------------- dense GEMM: XW = X @ W (bf16 WMMA, f32 accum) ----------------
// One wave per 16-row strip; TILES_N accumulators reuse each A fragment across
// TILES_N back-to-back WMMAs. Requires M % 16 == 0 (true: 50000 = 3125*16).
template <int TILES_N>
__global__ void gcn_gemm_wmma(const float* __restrict__ X,
                              const __bf16* __restrict__ Wpk,
                              float* __restrict__ XW,
                              int tiles_m, int K, int relu_in)
{
    int wave = blockIdx.x * (blockDim.x >> 5) + (threadIdx.x >> 5);
    if (wave >= tiles_m) return;               // wave-uniform exit
    int lane = threadIdx.x & 31;
    int half = lane >> 4;
    int m0   = wave << 4;
    int row  = m0 + (lane & 15);
    constexpr int Nout = TILES_N * 16;

    f32x8 acc[TILES_N];
#pragma unroll
    for (int t = 0; t < TILES_N; ++t) acc[t] = (f32x8){};

    int kblocks = K >> 5;
    for (int kb = 0; kb < kblocks; ++kb) {
        // A fragment: two contiguous 8-float spans -> four global_load_b128
        const float4* xr = (const float4*)(X + (size_t)row * K + (kb << 5) + (half << 3));
        float4 s0 = xr[0], s1 = xr[1];         // K = 32kb + 8h + 0..7
        float4 s2 = xr[4], s3 = xr[5];         // K = 32kb + 16 + 8h + 0..7
        if (relu_in) { s0 = relu4(s0); s1 = relu4(s1); s2 = relu4(s2); s3 = relu4(s3); }
        bf16x16 a;
        a[0] = to_bf16(s0.x); a[1] = to_bf16(s0.y); a[2] = to_bf16(s0.z); a[3] = to_bf16(s0.w);
        a[4] = to_bf16(s1.x); a[5] = to_bf16(s1.y); a[6] = to_bf16(s1.z); a[7] = to_bf16(s1.w);
        a[8] = to_bf16(s2.x); a[9] = to_bf16(s2.y); a[10] = to_bf16(s2.z); a[11] = to_bf16(s2.w);
        a[12] = to_bf16(s3.x); a[13] = to_bf16(s3.y); a[14] = to_bf16(s3.z); a[15] = to_bf16(s3.w);
#pragma unroll
        for (int t = 0; t < TILES_N; ++t) {
            bf16x16 b = *(const bf16x16*)(Wpk + ((size_t)(kb * TILES_N + t) << 9) + (lane << 4));
            acc[t] = __builtin_amdgcn_wmma_f32_16x16x32_bf16(
                         false, a, false, b, (short)0, acc[t], false, false);
        }
    }
#pragma unroll
    for (int t = 0; t < TILES_N; ++t) {
#pragma unroll
        for (int r = 0; r < 8; ++r) {          // C layout: VGPR r -> M = r + 8*half
            int orow = m0 + r + (half << 3);
            XW[(size_t)orow * Nout + (t << 4) + (lane & 15)] = acc[t][r];
        }
    }
}

// ---------------- epilogue init: AGG = XW * dinv^2 + b ----------------
__global__ void gcn_init_agg(float* __restrict__ AGG, const float* __restrict__ XW,
                             const float* __restrict__ dinv, const float* __restrict__ bias,
                             int total, int shift)
{
    int i = blockIdx.x * blockDim.x + threadIdx.x;
    if (i >= total) return;
    int node = i >> shift;
    int f    = i & ((1 << shift) - 1);
    float di = dinv[node];
    AGG[i] = XW[i] * di * di + bias[f];
}

// ---------------- edge scatter: AGG[dst] += dinv[s]*w*dinv[d] * XW[src] ----------------
// One wave32 per edge; lane handles NOUT/32 consecutive features (vector load + atomics).
template <int NOUT>
__global__ void gcn_edge_scatter(float* __restrict__ AGG, const float* __restrict__ XW,
                                 const long long* __restrict__ src,
                                 const long long* __restrict__ dst,
                                 const float* __restrict__ ew,
                                 const float* __restrict__ dinv,
                                 int E)
{
    int e = blockIdx.x * (blockDim.x >> 5) + (threadIdx.x >> 5);
    if (e >= E) return;
    int lane = threadIdx.x & 31;
    int s = (int)src[e], d = (int)dst[e];
    float c = dinv[s] * ew[e] * dinv[d];
    constexpr int PER = NOUT / 32;             // 4 (128) or 2 (64)
    const float* xs = XW  + (size_t)s * NOUT + lane * PER;
    float*       ad = AGG + (size_t)d * NOUT + lane * PER;
    float v[PER];
#pragma unroll
    for (int j = 0; j < PER; ++j) v[j] = xs[j];   // merges to b128/b64 load
#pragma unroll
    for (int j = 0; j < PER; ++j) atomicAdd(ad + j, c * v[j]);
}

// ---------------- driver ----------------
extern "C" void kernel_launch(void* const* d_in, const int* in_sizes, int n_in,
                              void* d_out, int out_size, void* d_ws, size_t ws_size,
                              hipStream_t stream) {
    const float*     x   = (const float*)d_in[0];
    const long long* ei  = (const long long*)d_in[1];   // int64 [2, E]
    const float*     ew  = (const float*)d_in[2];
    const float* Wl[4] = {(const float*)d_in[3], (const float*)d_in[5],
                          (const float*)d_in[7], (const float*)d_in[9]};
    const float* bl[4] = {(const float*)d_in[4], (const float*)d_in[6],
                          (const float*)d_in[8], (const float*)d_in[10]};
    float* out = (float*)d_out;

    const int sizes[5] = {128, 128, 128, 128, 64};
    const int N = in_sizes[0] / sizes[0];
    const int E = in_sizes[2];
    const long long* srcp = ei;
    const long long* dstp = ei + E;

    // workspace: dinv | Wpk (bf16 fragments, <=32KB) | bufA (xw) | bufB (agg/x)
    char*   ws   = (char*)d_ws;
    float*  dinv = (float*)ws;
    size_t  off  = ((size_t)N * sizeof(float) + 511) & ~(size_t)511;
    __bf16* Wpk  = (__bf16*)(ws + off);
    off         += (128 * 128 * sizeof(__bf16) + 511) & ~(size_t)511;
    float*  bufA = (float*)(ws + off);
    off         += (size_t)N * 128 * sizeof(float);
    float*  bufB = (float*)(ws + off);

    gcn_fill_deg <<<(N + 255) / 256, 256, 0, stream>>>(dinv, N);
    gcn_accum_deg<<<(E + 255) / 256, 256, 0, stream>>>(dinv, dstp, ew, E);
    gcn_make_dinv<<<(N + 255) / 256, 256, 0, stream>>>(dinv, N);

    const float* X = x;
    const int tiles_m = N / 16;                // 50000 % 16 == 0
    for (int l = 0; l < 4; ++l) {
        int K = sizes[l], OUT = sizes[l + 1];
        float* AGG = (l < 3) ? bufB : out;
        int relu = (l > 0) ? 1 : 0;

        int pack_waves = (OUT >> 4) * (K >> 5);
        gcn_pack_w<<<(pack_waves + 3) / 4, 128, 0, stream>>>(Wl[l], Wpk, K, OUT);

        if (OUT == 128)
            gcn_gemm_wmma<8><<<(tiles_m + 3) / 4, 128, 0, stream>>>(X, Wpk, bufA, tiles_m, K, relu);
        else
            gcn_gemm_wmma<4><<<(tiles_m + 3) / 4, 128, 0, stream>>>(X, Wpk, bufA, tiles_m, K, relu);

        int tot = N * OUT;
        int shift = (OUT == 128) ? 7 : 6;
        gcn_init_agg<<<(tot + 255) / 256, 256, 0, stream>>>(AGG, bufA, dinv, bl[l], tot, shift);

        if (OUT == 128)
            gcn_edge_scatter<128><<<(E + 3) / 4, 128, 0, stream>>>(AGG, bufA, srcp, dstp, ew, dinv, E);
        else
            gcn_edge_scatter<64><<<(E + 3) / 4, 128, 0, stream>>>(AGG, bufA, srcp, dstp, ew, dinv, E);

        X = AGG;   // next layer input (ReLU fused into next GEMM's A load)
    }
}
